// MultiHeadAttention_73607149519274
// MI455X (gfx1250) — compile-verified, matrix operand order checked by reading, not run on hardware
//
#include <hip/hip_runtime.h>
#include <hip/hip_bf16.h>

// MHA forward for MI455X (gfx1250, wave32).
// All GEMMs run through v_wmma_f32_16x16x32_bf16 (fp32 accumulate).
// Shapes: B=2, S=2048, D=768, H=12, DK=64.

#define S_LEN   2048
#define D_MODEL 768
#define N_HEADS 12
#define D_K     64
#define M_ROWS  4096   // B*S

typedef __attribute__((ext_vector_type(16))) __bf16 v16bf;
typedef __attribute__((ext_vector_type(8)))  float  v8f;
typedef __attribute__((ext_vector_type(4)))  unsigned int u32x4;

// ---------- helpers ----------

static __device__ __forceinline__ unsigned short f2bf(float f) {
    unsigned u = __float_as_uint(f);
    u += 0x7FFFu + ((u >> 16) & 1u);          // round-to-nearest-even
    return (unsigned short)(u >> 16);
}

// A-style (and B-style) bf16 fragment load for v_wmma_f32_16x16x32_bf16.
// ISA layout (16-bit A, 16x32): lane&15 = row; klo = (lane&16)?8:0;
// dwords 0..3 = K[klo..klo+7], dwords 4..7 = K[16+klo..16+klo+7].
// `p` points at the fragment origin (row 0, k 0); rows are `stride` ushorts apart.
// Stride must be a multiple of 8 so the two 16B chunks stay aligned.
static __device__ __forceinline__ v16bf frag_ld(const unsigned short* p, int stride) {
    const int lane = threadIdx.x & 31;
    const unsigned short* r = p + (lane & 15) * stride + ((lane & 16) ? 8 : 0);
    v16bf f;
    u32x4* h = (u32x4*)&f;
    h[0] = *(const u32x4*)(r);
    h[1] = *(const u32x4*)(r + 16);
    return f;
}

static __device__ __forceinline__ v8f wmma_bf16(v16bf a, v16bf b, v8f c) {
    return __builtin_amdgcn_wmma_f32_16x16x32_bf16(
        /*neg_a=*/false, a, /*neg_b=*/false, b,
        /*c_mod=*/(short)0, c, /*reuse_a=*/false, /*reuse_b=*/false);
}

// ---------- fp32 -> bf16 conversion (vectorized) ----------

__global__ __launch_bounds__(256) void cvt_bf16_kernel(
    const float* __restrict__ src, unsigned short* __restrict__ dst, int n4) {
    int i = blockIdx.x * blockDim.x + threadIdx.x;
    if (i < n4) {
        float4 f = ((const float4*)src)[i];
        uint2 o;
        o.x = (unsigned)f2bf(f.x) | ((unsigned)f2bf(f.y) << 16);
        o.y = (unsigned)f2bf(f.z) | ((unsigned)f2bf(f.w) << 16);
        ((uint2*)dst)[i] = o;
    }
}

// ---------- bf16 GEMM: Y[m,n] = sum_k X[m,k]*W[n,k] + bias[n] ----------
// K = N = 768 fixed. Block tile 128(M) x 128(N), BK=64, 256 threads = 8 waves.
// Waves in a 4(M) x 2(N) grid; each wave owns a 32x64 tile = 2x4 WMMA frags
// -> 8 WMMAs per 12 ds_load_b128 per 32-K substep (1.5 LDS loads / WMMA).
// mode 0: bf16 head-major   [B,H,S,DK]   (Q/K projections)
// mode 2: bf16 head-transp. [B,H,DK,S]   (V projection -> transpose-free flash)
// mode 1: fp32 row-major    [M,768]      (final projection -> d_out)

#define LDG 72   // padded LDS row stride (144B, 16B aligned, conflict-skewed)

__global__ __launch_bounds__(256) void gemm_bf16_kernel(
    const unsigned short* __restrict__ X, const unsigned short* __restrict__ W,
    const float* __restrict__ bias, void* __restrict__ Out, int mode) {
    __shared__ unsigned short Xs[128 * LDG];
    __shared__ unsigned short Ws[128 * LDG];

    const int t    = threadIdx.x;
    const int lane = t & 31;
    const int wave = t >> 5;
    const int wy   = wave >> 1;     // 0..3  (M)
    const int wx   = wave & 1;      // 0..1  (N)
    const int blockM = blockIdx.x * 128;
    const int blockN = blockIdx.y * 128;

    v8f c[2][4] = {};

    for (int k0 = 0; k0 < D_MODEL; k0 += 64) {
        // Stage X tile 128x64 and W tile 128x64 (each thread: 4+4 b128 loads)
        {
            int row = t >> 1, cb = (t & 1) * 32;
            const u32x4* gx = (const u32x4*)(X + (size_t)(blockM + row) * D_MODEL + k0 + cb);
            u32x4* sx = (u32x4*)(Xs + row * LDG + cb);
            sx[0] = gx[0]; sx[1] = gx[1]; sx[2] = gx[2]; sx[3] = gx[3];

            const u32x4* gw = (const u32x4*)(W + (size_t)(blockN + row) * D_MODEL + k0 + cb);
            u32x4* sw = (u32x4*)(Ws + row * LDG + cb);
            sw[0] = gw[0]; sw[1] = gw[1]; sw[2] = gw[2]; sw[3] = gw[3];
        }
        __syncthreads();

#pragma unroll
        for (int kk = 0; kk < 64; kk += 32) {
            v16bf a0 = frag_ld(Xs + (wy * 32 + 0)  * LDG + kk, LDG);
            v16bf a1 = frag_ld(Xs + (wy * 32 + 16) * LDG + kk, LDG);
#pragma unroll
            for (int j = 0; j < 4; ++j) {
                v16bf b = frag_ld(Ws + (wx * 64 + j * 16) * LDG + kk, LDG);
                c[0][j] = wmma_bf16(a0, b, c[0][j]);
                c[1][j] = wmma_bf16(a1, b, c[1][j]);
            }
        }
        __syncthreads();
    }

    // Epilogue. C layout: element r of lane -> row = r + 8*(lane>>4), col = lane&15.
#pragma unroll
    for (int i = 0; i < 2; ++i)
#pragma unroll
        for (int j = 0; j < 4; ++j) {
            int col = blockN + wx * 64 + j * 16 + (lane & 15);
            float bv = bias[col];
#pragma unroll
            for (int r = 0; r < 8; ++r) {
                int row = blockM + wy * 32 + i * 16 + r + ((lane >> 4) << 3);
                float v = c[i][j][r] + bv;
                int b  = row >> 11, s = row & 2047;
                int h  = col >> 6,  dk = col & 63;
                if (mode == 0) {        // [B,H,S,DK]
                    ((unsigned short*)Out)[(((size_t)(b * N_HEADS + h)) * S_LEN + s) * D_K + dk] = f2bf(v);
                } else if (mode == 2) { // [B,H,DK,S]  (transposed V)
                    ((unsigned short*)Out)[(((size_t)(b * N_HEADS + h)) * D_K + dk) * S_LEN + s] = f2bf(v);
                } else {                // fp32 [M, D_MODEL]
                    ((float*)Out)[(size_t)row * D_MODEL + col] = v;
                }
            }
        }
}

// ---------- flash attention ----------
// Grid: (S/128, B*H). 256 threads = 8 waves; wave w owns query rows [16w,16w+16).
// Key tile = 64 keys. K staged [key][dk]; V is already transposed in global
// ([B,H,DK,S]) so its tile stages with coalesced b128 rows too — no scalar
// transpose. Both score and ctx B-fragments load with the row-major A pattern.

#define LDK 72   // 144B rows: 16B aligned, conflict-skewed

__global__ __launch_bounds__(256) void flash_attn_kernel(
    const unsigned short* __restrict__ Q, const unsigned short* __restrict__ K,
    const unsigned short* __restrict__ Vt, unsigned short* __restrict__ Ctx) {
    __shared__ unsigned short Ks [64 * LDK];   // [key][dk]
    __shared__ unsigned short Vts[64 * LDK];   // [dk][key]
    __shared__ unsigned short Pb [8 * 16 * LDK];

    const int t = threadIdx.x, lane = t & 31, wave = t >> 5;
    const int bh    = blockIdx.y;                 // b*H + h
    const int qBase = blockIdx.x * 128 + wave * 16;
    const size_t headOff = (size_t)bh * S_LEN * D_K;

    // Q fragments straight from global (row stride = 64 bf16 = 128B, 16B aligned)
    const unsigned short* qp = Q + headOff + (size_t)qBase * D_K;
    v16bf qa0 = frag_ld(qp,      D_K);   // dk  0..31
    v16bf qa1 = frag_ld(qp + 32, D_K);   // dk 32..63

    float m[8], l[8];
#pragma unroll
    for (int r = 0; r < 8; ++r) { m[r] = -1e30f; l[r] = 0.0f; }
    v8f o[4] = {};

    for (int kt = 0; kt < S_LEN; kt += 64) {
        // Stage K tile [key][dk] and Vt tile [dk][key]: 4 threads per row.
        {
            int row = t >> 2, qd = (t & 3) * 16;
            const u32x4* gk = (const u32x4*)(K + headOff + (size_t)(kt + row) * D_K + qd);
            u32x4* sk = (u32x4*)(Ks + row * LDK + qd);
            sk[0] = gk[0]; sk[1] = gk[1];

            const u32x4* gv = (const u32x4*)(Vt + headOff + (size_t)row * S_LEN + kt + qd);
            u32x4* sv = (u32x4*)(Vts + row * LDK + qd);
            sv[0] = gv[0]; sv[1] = gv[1];

            if (kt + 64 < S_LEN) {  // prefetch next tile -> global_prefetch_b8
                __builtin_prefetch(K  + headOff + (size_t)(kt + 64 + row) * D_K + qd, 0, 1);
                __builtin_prefetch(Vt + headOff + (size_t)row * S_LEN + kt + 64 + qd, 0, 1);
            }
        }
        __syncthreads();

        // scores S = Q (16x64) @ K_tile^T -> 4 n-tiles of 16 keys
        v8f sc[4];
#pragma unroll
        for (int j = 0; j < 4; ++j) {
            v16bf b0 = frag_ld(Ks + (j * 16) * LDK,      LDK);  // dk  0..31
            v16bf b1 = frag_ld(Ks + (j * 16) * LDK + 32, LDK);  // dk 32..63
            v8f s0 = {};
            s0 = wmma_bf16(qa0, b0, s0);
            s0 = wmma_bf16(qa1, b1, s0);
            sc[j] = s0;
        }

        // online softmax (rows live per 16-lane half: reduce with masks 1,2,4,8)
        const float scale = 0.125f;  // 1/sqrt(64)
        float mnew[8], alpha[8], rsum[8];
#pragma unroll
        for (int r = 0; r < 8; ++r) {
            float v = sc[0][r] * scale;
            v = fmaxf(v, sc[1][r] * scale);
            v = fmaxf(v, sc[2][r] * scale);
            v = fmaxf(v, sc[3][r] * scale);
#pragma unroll
            for (int msk = 1; msk < 16; msk <<= 1)
                v = fmaxf(v, __shfl_xor(v, msk, 32));
            mnew[r]  = fmaxf(m[r], v);
            alpha[r] = __expf(m[r] - mnew[r]);
            m[r]     = mnew[r];
            rsum[r]  = 0.0f;
        }
#pragma unroll
        for (int j = 0; j < 4; ++j)
#pragma unroll
            for (int r = 0; r < 8; ++r) {
                float p = __expf(sc[j][r] * scale - mnew[r]);
                sc[j][r] = p;
                rsum[r] += p;
            }
#pragma unroll
        for (int r = 0; r < 8; ++r) {
#pragma unroll
            for (int msk = 1; msk < 16; msk <<= 1)
                rsum[r] += __shfl_xor(rsum[r], msk, 32);
            l[r] = l[r] * alpha[r] + rsum[r];
        }

        // C-layout -> A-fragment layout via per-wave LDS region (no barrier needed:
        // DS ops are in-order within a wave)
        unsigned short* pbase = Pb + wave * 16 * LDK;
#pragma unroll
        for (int j = 0; j < 4; ++j)
#pragma unroll
            for (int r = 0; r < 8; ++r)
                pbase[(r + ((lane >> 4) << 3)) * LDK + j * 16 + (lane & 15)] = f2bf(sc[j][r]);
        v16bf pa0 = frag_ld(pbase,      LDK);  // keys  0..31
        v16bf pa1 = frag_ld(pbase + 32, LDK);  // keys 32..63

        // rescale running output, then O += P @ V
#pragma unroll
        for (int j = 0; j < 4; ++j)
#pragma unroll
            for (int r = 0; r < 8; ++r)
                o[j][r] *= alpha[r];
#pragma unroll
        for (int j = 0; j < 4; ++j) {
            v16bf vb0 = frag_ld(Vts + (j * 16) * LDK,      LDK);  // keys  0..31, dk tile j
            v16bf vb1 = frag_ld(Vts + (j * 16) * LDK + 32, LDK);  // keys 32..63
            o[j] = wmma_bf16(pa0, vb0, o[j]);
            o[j] = wmma_bf16(pa1, vb1, o[j]);
        }
        __syncthreads();
    }

    // finalize: ctx[b, s, h*64+dk] = o / l   (bf16, row-major [B*S, 768])
    const int b = bh / N_HEADS, h = bh % N_HEADS;
#pragma unroll
    for (int j = 0; j < 4; ++j)
#pragma unroll
        for (int r = 0; r < 8; ++r) {
            int srow = qBase + r + ((lane >> 4) << 3);
            int dk   = j * 16 + (lane & 15);
            float v  = o[j][r] / l[r];
            Ctx[((size_t)b * S_LEN + srow) * D_MODEL + h * D_K + dk] = f2bf(v);
        }
}

// ---------- launch ----------

extern "C" void kernel_launch(void* const* d_in, const int* in_sizes, int n_in,
                              void* d_out, int out_size, void* d_ws, size_t ws_size,
                              hipStream_t stream) {
    const float* x   = (const float*)d_in[0];
    const float* w_q = (const float*)d_in[1];
    const float* b_q = (const float*)d_in[2];
    const float* w_k = (const float*)d_in[3];
    const float* b_k = (const float*)d_in[4];
    const float* w_v = (const float*)d_in[5];
    const float* b_v = (const float*)d_in[6];
    const float* w_o = (const float*)d_in[7];
    const float* b_o = (const float*)d_in[8];
    float* out = (float*)d_out;

    // Workspace carve-up (bf16/ushort elements), ~36.2 MB total.
    const size_t nX = (size_t)M_ROWS * D_MODEL;     // 3,145,728
    const size_t nW = (size_t)D_MODEL * D_MODEL;    //   589,824
    unsigned short* ws   = (unsigned short*)d_ws;
    unsigned short* xb   = ws;
    unsigned short* wqb  = xb  + nX;
    unsigned short* wkb  = wqb + nW;
    unsigned short* wvb  = wkb + nW;
    unsigned short* wob  = wvb + nW;
    unsigned short* qb   = wob + nW;
    unsigned short* kb   = qb  + nX;
    unsigned short* vtb  = kb  + nX;   // V stored transposed [B,H,DK,S]
    unsigned short* ctxb = vtb + nX;

    // 1) fp32 -> bf16 conversions
    {
        int n4 = (int)(nX / 4);
        cvt_bf16_kernel<<<(n4 + 255) / 256, 256, 0, stream>>>(x, xb, n4);
        int w4 = (int)(nW / 4);
        cvt_bf16_kernel<<<(w4 + 255) / 256, 256, 0, stream>>>(w_q, wqb, w4);
        cvt_bf16_kernel<<<(w4 + 255) / 256, 256, 0, stream>>>(w_k, wkb, w4);
        cvt_bf16_kernel<<<(w4 + 255) / 256, 256, 0, stream>>>(w_v, wvb, w4);
        cvt_bf16_kernel<<<(w4 + 255) / 256, 256, 0, stream>>>(w_o, wob, w4);
    }

    // 2) QKV projections (bf16 WMMA); V goes out pre-transposed for flash
    dim3 ggrid(M_ROWS / 128, D_MODEL / 128);  // 32 x 6
    gemm_bf16_kernel<<<ggrid, 256, 0, stream>>>(xb, wqb, b_q, qb,  0);
    gemm_bf16_kernel<<<ggrid, 256, 0, stream>>>(xb, wkb, b_k, kb,  0);
    gemm_bf16_kernel<<<ggrid, 256, 0, stream>>>(xb, wvb, b_v, vtb, 2);

    // 3) flash attention -> ctx (bf16 row-major)
    dim3 agrid(S_LEN / 128, 2 * N_HEADS);     // 16 x 24
    flash_attn_kernel<<<agrid, 256, 0, stream>>>(qb, kb, vtb, ctxb);

    // 4) output projection -> fp32 d_out
    gemm_bf16_kernel<<<ggrid, 256, 0, stream>>>(ctxb, wob, b_o, out, 1);
}
